// HomogeneousDeepGraphEmbedding2_76725295775978
// MI455X (gfx1250) — compile-verified, matrix operand order checked by reading, not run on hardware
//
#include <hip/hip_runtime.h>
#include <cmath>

#define FDIM 128
#define LRELU_SLOPE 0.2f
#define BN_EPS 1e-5f
#define SASTR 136   // LDS row stride in halves (128 + 8 pad -> conflict-free fragment reads)

typedef __attribute__((ext_vector_type(16))) _Float16 v16h;
typedef __attribute__((ext_vector_type(8)))  _Float16 v8h;
typedef __attribute__((ext_vector_type(4)))  _Float16 v4h;
typedef __attribute__((ext_vector_type(8)))  float    v8f;
typedef __attribute__((ext_vector_type(4)))  float    v4f;

// ---------------- wave helpers ----------------
__device__ __forceinline__ float wred_sum32(float v) {
#pragma unroll
  for (int o = 16; o > 0; o >>= 1) v += __shfl_xor(v, o, 32);
  return v;
}
__device__ __forceinline__ float wred_sum16(float v) {  // reduce within half-wave (16 lanes)
#pragma unroll
  for (int o = 8; o > 0; o >>= 1) v += __shfl_xor(v, o, 32);
  return v;
}

// monotone float <-> uint encoding for atomic max on floats of any sign
__device__ __forceinline__ unsigned fenc(float x) {
  unsigned u = __float_as_uint(x);
  return (u & 0x80000000u) ? ~u : (u | 0x80000000u);
}
__device__ __forceinline__ float fdec(unsigned u) {
  u = (u & 0x80000000u) ? (u & 0x7FFFFFFFu) : ~u;
  return __uint_as_float(u);
}

// ---------------- weight pre-transpose+convert: Wt[n][k] (f16) = W[k][n] (f32) ----------------
__global__ void k_cvt_wt(const float* __restrict__ W, _Float16* __restrict__ Wt) {
  int i = blockIdx.x * blockDim.x + threadIdx.x;   // element-quad index
  int t = i * 4;
  if (t >= FDIM * FDIM) return;
  int n = t >> 7, k0 = t & (FDIM - 1);
  v4h o;
#pragma unroll
  for (int j = 0; j < 4; ++j) o[j] = (_Float16)W[(size_t)(k0 + j) * FDIM + n];
  *(v4h*)&Wt[t] = o;
}

// ---------------- WMMA GEMM: C[M x 128] = A[M x 128] * W[128 x 128] (+bias) ----------------
// Block: 128 threads (4 waves), 16 rows of A staged in LDS as f16.
// Each wave owns two 16-col tiles -> two independent WMMA accumulator chains.
__global__ void __launch_bounds__(128) k_gemm128(const float* __restrict__ A,
                                                 const _Float16* __restrict__ Wt, // [n][k] f16
                                                 const float* __restrict__ bias,
                                                 float* __restrict__ C, int M) {
  __shared__ __align__(16) _Float16 sA[16 * SASTR];
  int t = threadIdx.x;
  int lane = t & 31;
  int wave = t >> 5;                 // 0..3
  int row0 = blockIdx.x << 4;
  if (row0 >= M) return;             // uniform
  {  // cooperative load+convert of 16x128 A tile (each thread: 16 floats -> 16 halves)
    int r = t >> 3;                  // 0..15
    int q = t & 7;                   // cols q*16 .. q*16+15
    const float* p = A + (size_t)(row0 + r) * FDIM + q * 16;
    v4f f0 = *(const v4f*)(p);
    v4f f1 = *(const v4f*)(p + 4);
    v4f f2 = *(const v4f*)(p + 8);
    v4f f3 = *(const v4f*)(p + 12);
    v8h h0, h1;
#pragma unroll
    for (int j = 0; j < 4; ++j) {
      h0[j] = (_Float16)f0[j]; h0[4 + j] = (_Float16)f1[j];
      h1[j] = (_Float16)f2[j]; h1[4 + j] = (_Float16)f3[j];
    }
    *(v8h*)&sA[r * SASTR + q * 16]     = h0;
    *(v8h*)&sA[r * SASTR + q * 16 + 8] = h1;
  }
  __syncthreads();
  int m  = lane & 15;
  int na = (wave << 5) + m;          // first col tile column
  int nb = na + 16;                  // second col tile column
  v8f accA = {}, accB = {};
#pragma unroll
  for (int kt = 0; kt < 4; ++kt) {
    int kb = kt * 32 + ((lane >> 4) << 3);  // ISA 7.12.2 K striping
    v8h alo = *(const v8h*)&sA[m * SASTR + kb];
    v8h ahi = *(const v8h*)&sA[m * SASTR + kb + 16];
    v16h a;
#pragma unroll
    for (int j = 0; j < 8; ++j) { a[j] = alo[j]; a[8 + j] = ahi[j]; }
    v8h blo = *(const v8h*)&Wt[(size_t)na * FDIM + kb];
    v8h bhi = *(const v8h*)&Wt[(size_t)na * FDIM + kb + 16];
    v16h b0;
#pragma unroll
    for (int j = 0; j < 8; ++j) { b0[j] = blo[j]; b0[8 + j] = bhi[j]; }
    v8h clo = *(const v8h*)&Wt[(size_t)nb * FDIM + kb];
    v8h chi = *(const v8h*)&Wt[(size_t)nb * FDIM + kb + 16];
    v16h b1;
#pragma unroll
    for (int j = 0; j < 8; ++j) { b1[j] = clo[j]; b1[8 + j] = chi[j]; }
    accA = __builtin_amdgcn_wmma_f32_16x16x32_f16(false, a, false, b0, (short)0, accA, false, false);
    accB = __builtin_amdgcn_wmma_f32_16x16x32_f16(false, a, false, b1, (short)0, accB, false, false);
  }
  int rbase = row0 + ((lane >> 4) << 3);
  float bva = bias ? bias[na] : 0.f;
  float bvb = bias ? bias[nb] : 0.f;
#pragma unroll
  for (int r = 0; r < 8; ++r) {
    C[(size_t)(rbase + r) * FDIM + na] = accA[r] + bva;
    C[(size_t)(rbase + r) * FDIM + nb] = accB[r] + bvb;
  }
}

// ---------------- fills (float4 granular) ----------------
__global__ void k_fill4_f32(v4f* __restrict__ p, float v, long long n4) {
  long long i = (long long)blockIdx.x * blockDim.x + threadIdx.x;
  if (i < n4) { v4f o = {v, v, v, v}; p[i] = o; }
}
__global__ void k_fill_u32(unsigned* __restrict__ p, unsigned v, long long n) {
  long long i = (long long)blockIdx.x * blockDim.x + threadIdx.x;
  if (i < n) p[i] = v;
}

// ---------------- elementwise (float4 per thread) ----------------
__global__ void k_bias_bn_relu(const float* __restrict__ in, const float* __restrict__ colbias,
                               const float* __restrict__ g, const float* __restrict__ b,
                               const float* __restrict__ m, const float* __restrict__ v,
                               float* __restrict__ out, long long total4) {
  long long i = (long long)blockIdx.x * blockDim.x + threadIdx.x;
  if (i >= total4) return;
  long long base = i * 4;
  int c0 = (int)(base & (FDIM - 1));
  v4f xv = *(const v4f*)(in + base);
  v4f gv = *(const v4f*)(g + c0), bv = *(const v4f*)(b + c0);
  v4f mv = *(const v4f*)(m + c0), vv = *(const v4f*)(v + c0);
#pragma unroll
  for (int j = 0; j < 4; ++j) {
    float x = xv[j] + (colbias ? colbias[c0 + j] : 0.f);
    x = (x - mv[j]) * rsqrtf(vv[j] + BN_EPS) * gv[j] + bv[j];
    xv[j] = x > 0.f ? x : 0.f;
  }
  *(v4f*)(out + base) = xv;
}

__global__ void k_axpby(const float* __restrict__ X, const float* __restrict__ Y,
                        float a, float b, float* __restrict__ out, long long total4) {
  long long i = (long long)blockIdx.x * blockDim.x + threadIdx.x;
  if (i >= total4) return;
  long long base = i * 4;
  v4f xv = *(const v4f*)(X + base);
  v4f yv = *(const v4f*)(Y + base);
#pragma unroll
  for (int j = 0; j < 4; ++j) xv[j] = a * xv[j] + b * yv[j];
  *(v4f*)(out + base) = xv;
}

// ---------------- GATv2 edge kernels: wave/edge, lane owns 4 contiguous channels ----------------
// heads=2 x 64ch: head = lane>>4 ; att flat index att[h*64+cc] == att[c]
__global__ void __launch_bounds__(256) k_gat_logit(const int* __restrict__ ei, int E, long long ETOT,
                                                   const float* __restrict__ xl, const float* __restrict__ xr,
                                                   const float* __restrict__ att,
                                                   float* __restrict__ logit, unsigned* __restrict__ menc) {
  int lane = threadIdx.x & 31;
  long long eid = (long long)blockIdx.x * 8 + (threadIdx.x >> 5);
  if (eid >= ETOT) return;
  int src, dst;
  if (eid < E) { src = ei[eid]; dst = ei[E + eid]; }
  else         { src = dst = (int)(eid - E); }
  int c0 = lane * 4;
  v4f a  = *(const v4f*)(xl + (size_t)src * FDIM + c0);
  v4f b  = *(const v4f*)(xr + (size_t)dst * FDIM + c0);
  v4f aw = *(const v4f*)(att + c0);
  float s = 0.f;
#pragma unroll
  for (int j = 0; j < 4; ++j) {
    float z = a[j] + b[j];
    z = z > 0.f ? z : LRELU_SLOPE * z;
    s += z * aw[j];
  }
  s = wred_sum16(s);                     // per-head partial (halves don't mix)
  if ((lane & 15) == 0) {
    int h = lane >> 4;
    logit[eid * 2 + h] = s;
    atomicMax(&menc[(size_t)dst * 2 + h], fenc(s));
  }
}

__global__ void k_gat_expsum(const int* __restrict__ ei, int E, long long ETOT,
                             const unsigned* __restrict__ menc,
                             float* __restrict__ logit,  // in: e  out: p
                             float* __restrict__ denom) {
  long long i = (long long)blockIdx.x * blockDim.x + threadIdx.x;
  if (i >= ETOT * 2) return;
  long long eid = i >> 1;
  int h = (int)(i & 1);
  int dst = (eid < E) ? ei[E + eid] : (int)(eid - E);
  float mx = fdec(menc[(size_t)dst * 2 + h]);
  float p  = __expf(logit[i] - mx);
  logit[i] = p;
  atomicAdd(&denom[(size_t)dst * 2 + h], p);
}

__global__ void __launch_bounds__(256) k_gat_agg(const int* __restrict__ ei, int E, long long ETOT,
                                                 const float* __restrict__ xl,
                                                 const float* __restrict__ p, const float* __restrict__ denom,
                                                 float* __restrict__ out) {
  int lane = threadIdx.x & 31;
  long long eid = (long long)blockIdx.x * 8 + (threadIdx.x >> 5);
  if (eid >= ETOT) return;
  int src, dst;
  if (eid < E) { src = ei[eid]; dst = ei[E + eid]; }
  else         { src = dst = (int)(eid - E); }
  int h = lane >> 4;
  float alpha = p[eid * 2 + h] / denom[(size_t)dst * 2 + h];
  int c0 = lane * 4;
  v4f s = *(const v4f*)(xl + (size_t)src * FDIM + c0);
  float* po = out + (size_t)dst * FDIM + c0;
#pragma unroll
  for (int j = 0; j < 4; ++j) atomicAdd(&po[j], alpha * s[j]);
}

// ---------------- GCN norm + propagation ----------------
__global__ void k_deg(const int* __restrict__ ei, const float* __restrict__ ew, int E,
                      float* __restrict__ deg) {
  long long i = (long long)blockIdx.x * blockDim.x + threadIdx.x;
  if (i >= E) return;
  atomicAdd(&deg[ei[E + i]], ew[i]);
}
__global__ void k_rsqrt_inplace(float* __restrict__ d, int N) {
  long long i = (long long)blockIdx.x * blockDim.x + threadIdx.x;
  if (i >= N) return;
  float v = d[i];
  d[i] = v > 0.f ? rsqrtf(v) : 0.f;
}
__global__ void k_prop_init(const float* __restrict__ x, const float* __restrict__ dinv,
                            float* __restrict__ out, long long total4) {
  long long i = (long long)blockIdx.x * blockDim.x + threadIdx.x;
  if (i >= total4) return;
  long long base = i * 4;
  int n = (int)(base >> 7);
  float di = dinv[n];
  float w = di * di;
  v4f xv = *(const v4f*)(x + base);
#pragma unroll
  for (int j = 0; j < 4; ++j) xv[j] *= w;
  *(v4f*)(out + base) = xv;
}
__global__ void __launch_bounds__(256) k_prop(const int* __restrict__ ei, const float* __restrict__ ew, int E,
                                              const float* __restrict__ dinv,
                                              const float* __restrict__ x, float* __restrict__ out) {
  int lane = threadIdx.x & 31;
  long long eid = (long long)blockIdx.x * 8 + (threadIdx.x >> 5);
  if (eid >= E) return;
  int src = ei[eid], dst = ei[E + eid];
  float w = dinv[src] * ew[eid] * dinv[dst];
  int c0 = lane * 4;
  v4f s = *(const v4f*)(x + (size_t)src * FDIM + c0);
  float* po = out + (size_t)dst * FDIM + c0;
#pragma unroll
  for (int j = 0; j < 4; ++j) atomicAdd(&po[j], w * s[j]);
}

// ---------------- MemPooling (heads=4, clusters=2, tau=1), wave per node ----------------
__global__ void __launch_bounds__(256) k_mem(const float* __restrict__ x,
                                             const float* __restrict__ memk,   // [4][2][128]
                                             const float* __restrict__ convw,  // [4]
                                             int NPER, int N, float* __restrict__ pooled) {
  int lane = threadIdx.x & 31;
  long long n = (long long)blockIdx.x * 8 + (threadIdx.x >> 5);
  if (n >= N) return;
  int c0 = lane * 4;
  v4f xv = *(const v4f*)(x + (size_t)n * FDIM + c0);
  float xn2 = 0.f;
#pragma unroll
  for (int j = 0; j < 4; ++j) xn2 += xv[j] * xv[j];
  xn2 = wred_sum32(xn2);
  float s[4][2];
#pragma unroll
  for (int h = 0; h < 4; ++h) {
#pragma unroll
    for (int k = 0; k < 2; ++k) {
      v4f kv = *(const v4f*)(memk + (size_t)(h * 2 + k) * FDIM + c0);
      float dot = 0.f, kn = 0.f;
#pragma unroll
      for (int j = 0; j < 4; ++j) { dot += xv[j] * kv[j]; kn += kv[j] * kv[j]; }
      dot = wred_sum32(dot);
      kn  = wred_sum32(kn);
      float d2 = xn2 + kn - 2.f * dot;
      s[h][k] = 1.f / (1.f + d2);
    }
  }
  float t0 = 0.f, t1 = 0.f;
#pragma unroll
  for (int h = 0; h < 4; ++h) {
    float inv = 1.f / (s[h][0] + s[h][1]);
    t0 += convw[h] * s[h][0] * inv;
    t1 += convw[h] * s[h][1] * inv;
  }
  float mx = fmaxf(t0, t1);
  float e0 = __expf(t0 - mx), e1 = __expf(t1 - mx);
  float inv = 1.f / (e0 + e1);
  float S0 = e0 * inv, S1 = e1 * inv;
  int bg = (int)(n / NPER);
  float* pp = pooled + (size_t)bg * 2 * FDIM + c0;
#pragma unroll
  for (int j = 0; j < 4; ++j) {
    atomicAdd(&pp[j],        S0 * xv[j]);
    atomicAdd(&pp[FDIM + j], S1 * xv[j]);
  }
}

// pl[b][k*64+j] = sum_c pooled[b][k][c]*mlw[c][j] + mlb[j]
__global__ void k_memlin(const float* __restrict__ pooled, const float* __restrict__ mlw,
                         const float* __restrict__ mlb, float* __restrict__ pl, int B) {
  long long i = (long long)blockIdx.x * blockDim.x + threadIdx.x;
  if (i >= (long long)B * FDIM) return;
  int b = (int)(i >> 7);
  int r = (int)(i & (FDIM - 1));
  int k = r >> 6, j = r & 63;
  const float* pp = pooled + (size_t)b * 2 * FDIM + (size_t)k * FDIM;
  float acc = mlb[j];
#pragma unroll 4
  for (int c = 0; c < FDIM; ++c) acc += pp[c] * mlw[c * 64 + j];
  pl[i] = acc;
}

// out[b][o] = sum_t pl[b][t]*outw[t][o] + outb[o]
__global__ void k_out(const float* __restrict__ pl, const float* __restrict__ outw,
                      const float* __restrict__ outb, float* __restrict__ out, int B) {
  long long i = (long long)blockIdx.x * blockDim.x + threadIdx.x;
  if (i >= (long long)B * 8) return;
  int b = (int)(i >> 3), o = (int)(i & 7);
  const float* pp = pl + (size_t)b * FDIM;
  float acc = outb[o];
#pragma unroll 4
  for (int t = 0; t < FDIM; ++t) acc += pp[t] * outw[t * 8 + o];
  out[i] = acc;
}

// =================================================================================
extern "C" void kernel_launch(void* const* d_in, const int* in_sizes, int n_in,
                              void* d_out, int out_size, void* d_ws, size_t ws_size,
                              hipStream_t stream) {
  (void)n_in; (void)ws_size;
  const float* x     = (const float*)d_in[0];
  const int*   ei    = (const int*)d_in[1];
  const float* ew    = (const float*)d_in[2];
  const float* g1wl  = (const float*)d_in[3];
  const float* g1wr  = (const float*)d_in[4];
  const float* g1bl  = (const float*)d_in[5];
  const float* g1br  = (const float*)d_in[6];
  const float* g1att = (const float*)d_in[7];
  const float* g1bias= (const float*)d_in[8];
  const float* bn1g  = (const float*)d_in[9],  *bn1b = (const float*)d_in[10],
           *bn1m = (const float*)d_in[11], *bn1v = (const float*)d_in[12];
  const float* gcn2w = (const float*)d_in[13];
  const float* bn2g  = (const float*)d_in[14], *bn2b = (const float*)d_in[15],
           *bn2m = (const float*)d_in[16], *bn2v = (const float*)d_in[17];
  const float* gcnw  = (const float*)d_in[18];
  const float* gcnb  = (const float*)d_in[19];
  const float* bn3g  = (const float*)d_in[20], *bn3b = (const float*)d_in[21],
           *bn3m = (const float*)d_in[22], *bn3v = (const float*)d_in[23];
  const float* g2wl  = (const float*)d_in[24];
  const float* g2wr  = (const float*)d_in[25];
  const float* g2bl  = (const float*)d_in[26];
  const float* g2br  = (const float*)d_in[27];
  const float* g2att = (const float*)d_in[28];
  const float* g2bias= (const float*)d_in[29];
  const float* bn4g  = (const float*)d_in[30], *bn4b = (const float*)d_in[31],
           *bn4m = (const float*)d_in[32], *bn4v = (const float*)d_in[33];
  const float* memk  = (const float*)d_in[34];
  const float* convw = (const float*)d_in[35];
  const float* mlw   = (const float*)d_in[36];
  const float* mlb   = (const float*)d_in[37];
  const float* outw  = (const float*)d_in[38];
  const float* outb  = (const float*)d_in[39];

  const int  N    = in_sizes[0] / FDIM;     // 50000
  const int  E    = in_sizes[1] / 2;        // 500000
  const long long ETOT = (long long)E + N;  // with self loops
  const int  B    = out_size / 8;           // 50
  const int  NPER = N / B;                  // 1000
  const long long NB = (long long)N * FDIM;

  // workspace carve-out (all offsets stay 16B aligned)
  float*    ws     = (float*)d_ws;
  float*    buf0   = ws;                 // N*128
  float*    buf1   = buf0 + NB;          // N*128
  float*    buf2   = buf1 + NB;          // N*128
  float*    buf3   = buf2 + NB;          // N*128
  float*    logit  = buf3 + NB;          // ETOT*2
  unsigned* menc   = (unsigned*)(logit + ETOT * 2);   // N*2
  float*    denom  = (float*)(menc + (size_t)N * 2);  // N*2
  float*    dinv   = denom + (size_t)N * 2;           // N
  float*    pooled = dinv + N;                        // B*2*128
  float*    pl     = pooled + (size_t)B * 2 * FDIM;   // B*128
  _Float16* wtbase = (_Float16*)(pl + (size_t)B * FDIM);
  const size_t WSZ = (size_t)FDIM * FDIM;             // 16384 halves per matrix
  _Float16* wt_g1wl = wtbase + 0 * WSZ;
  _Float16* wt_g1wr = wtbase + 1 * WSZ;
  _Float16* wt_gcn2 = wtbase + 2 * WSZ;
  _Float16* wt_gcn  = wtbase + 3 * WSZ;
  _Float16* wt_g2wl = wtbase + 4 * WSZ;
  _Float16* wt_g2wr = wtbase + 5 * WSZ;

  auto cdiv = [](long long a, long long b) { return (unsigned)((a + b - 1) / b); };
  const unsigned gElem4   = cdiv(NB / 4, 256);
  const unsigned gWaveAll = cdiv(ETOT, 8);
  const unsigned gWaveE   = cdiv(E, 8);
  const unsigned gGemm    = cdiv(N, 16);
  const unsigned gCvt     = cdiv((long long)FDIM * FDIM / 4, 256);

  // ---- pre-transpose + f16-convert all six GEMM weights
  k_cvt_wt<<<gCvt, 256, 0, stream>>>(g1wl,  wt_g1wl);
  k_cvt_wt<<<gCvt, 256, 0, stream>>>(g1wr,  wt_g1wr);
  k_cvt_wt<<<gCvt, 256, 0, stream>>>(gcn2w, wt_gcn2);
  k_cvt_wt<<<gCvt, 256, 0, stream>>>(gcnw,  wt_gcn);
  k_cvt_wt<<<gCvt, 256, 0, stream>>>(g2wl,  wt_g2wl);
  k_cvt_wt<<<gCvt, 256, 0, stream>>>(g2wr,  wt_g2wr);

  auto gat_edges = [&](const float* xl, const float* xr, const float* att, float* agg) {
    k_fill_u32<<<cdiv(2LL * N, 256), 256, 0, stream>>>(menc, 0u, 2LL * N);
    k_fill4_f32<<<cdiv(2LL * N / 4, 256), 256, 0, stream>>>((v4f*)denom, 0.f, 2LL * N / 4);
    k_fill4_f32<<<cdiv(NB / 4, 256), 256, 0, stream>>>((v4f*)agg, 0.f, NB / 4);
    k_gat_logit<<<gWaveAll, 256, 0, stream>>>(ei, E, ETOT, xl, xr, att, logit, menc);
    k_gat_expsum<<<cdiv(ETOT * 2, 256), 256, 0, stream>>>(ei, E, ETOT, menc, logit, denom);
    k_gat_agg<<<gWaveAll, 256, 0, stream>>>(ei, E, ETOT, xl, logit, denom, agg);
  };

  // ---- GATv2 #1 : x4 = relu(bn1(gat(x))) -> buf2
  k_gemm128<<<gGemm, 128, 0, stream>>>(x, wt_g1wl, g1bl, buf0, N);
  k_gemm128<<<gGemm, 128, 0, stream>>>(x, wt_g1wr, g1br, buf1, N);
  gat_edges(buf0, buf1, g1att, buf2);
  k_bias_bn_relu<<<gElem4, 256, 0, stream>>>(buf2, g1bias, bn1g, bn1b, bn1m, bn1v, buf2, NB / 4);

  // ---- gcn_norm: dinv
  k_fill4_f32<<<cdiv((long long)N / 4, 256), 256, 0, stream>>>((v4f*)dinv, 1.0f, N / 4);  // self loop
  k_deg<<<cdiv(E, 256), 256, 0, stream>>>(ei, ew, E, dinv);
  k_rsqrt_inplace<<<cdiv(N, 256), 256, 0, stream>>>(dinv, N);

  // ---- GCN2Conv: hdd = prop(x); hdd = 0.5*hdd + 0.5*x4; x3 = (1-B)*hdd + B*(hdd@W)
  k_prop_init<<<gElem4, 256, 0, stream>>>(x, dinv, buf3, NB / 4);
  k_prop<<<gWaveE, 256, 0, stream>>>(ei, ew, E, dinv, x, buf3);
  k_axpby<<<gElem4, 256, 0, stream>>>(buf3, buf2, 0.5f, 0.5f, buf3, NB / 4);  // hdd
  k_gemm128<<<gGemm, 128, 0, stream>>>(buf3, wt_gcn2, nullptr, buf0, N);      // hdd @ gcn2_w
  const float BETA = logf(0.1f / 2.0f + 1.0f);
  k_axpby<<<gElem4, 256, 0, stream>>>(buf3, buf0, 1.f - BETA, BETA, buf1, NB / 4);
  k_bias_bn_relu<<<gElem4, 256, 0, stream>>>(buf1, nullptr, bn2g, bn2b, bn2m, bn2v, buf1, NB / 4);

  // ---- GCNConv: x3 = relu(bn3(prop(x3@gcn_w) + gcn_b)) -> buf2
  k_gemm128<<<gGemm, 128, 0, stream>>>(buf1, wt_gcn, nullptr, buf0, N);
  k_prop_init<<<gElem4, 256, 0, stream>>>(buf0, dinv, buf2, NB / 4);
  k_prop<<<gWaveE, 256, 0, stream>>>(ei, ew, E, dinv, buf0, buf2);
  k_bias_bn_relu<<<gElem4, 256, 0, stream>>>(buf2, gcnb, bn3g, bn3b, bn3m, bn3v, buf2, NB / 4);

  // ---- GATv2 #2 : x3 = relu(bn4(gat(x3))) -> buf3
  k_gemm128<<<gGemm, 128, 0, stream>>>(buf2, wt_g2wl, g2bl, buf0, N);
  k_gemm128<<<gGemm, 128, 0, stream>>>(buf2, wt_g2wr, g2br, buf1, N);
  gat_edges(buf0, buf1, g2att, buf3);
  k_bias_bn_relu<<<gElem4, 256, 0, stream>>>(buf3, g2bias, bn4g, bn4b, bn4m, bn4v, buf3, NB / 4);

  // ---- MemPooling + output head
  k_fill4_f32<<<cdiv((long long)B * 2 * FDIM / 4, 256), 256, 0, stream>>>((v4f*)pooled, 0.f, (long long)B * 2 * FDIM / 4);
  k_mem<<<cdiv(N, 8), 256, 0, stream>>>(buf3, memk, convw, NPER, N, pooled);
  k_memlin<<<cdiv((long long)B * FDIM, 256), 256, 0, stream>>>(pooled, mlw, mlb, pl, B);
  k_out<<<cdiv((long long)B * 8, 256), 256, 0, stream>>>(pl, outw, outb, (float*)d_out, B);
}
